// LI_26809185862268
// MI455X (gfx1250) — matrix operand order repaired and checked by reading, not compiled, not run on hardware
//
#include <hip/hip_runtime.h>

// ---- problem constants (from reference) ----
#define IN_F   1024
#define OUT_F  512
#define BATCH  32
#define TSTEPS 1024
#define MROWS  (BATCH * TSTEPS)          // 32768 GEMM rows
#define WELEMS (OUT_F * IN_F)            // 524288 weight elements

typedef __attribute__((ext_vector_type(16))) __bf16 v16bf;
typedef __attribute__((ext_vector_type(8)))  float  v8f;
typedef __attribute__((address_space(3))) __bf16 lds_bf16;

union F16x { float4 q[4]; float f[16]; };

// Split 16 fp32 values into bf16 hi + bf16 lo residual (Markidis split).
__device__ inline void split_bf16(const F16x& s, v16bf& hi, v16bf& lo) {
#pragma unroll
    for (int i = 0; i < 16; ++i) {
        float  v = s.f[i];
        __bf16 h = (__bf16)v;
        hi[i] = h;
        lo[i] = (__bf16)(v - (float)h);
    }
}

// 32-byte global -> LDS async copy (two b128 beats; INST_OFFSET advances both
// the memory and LDS addresses). Tracked by ASYNCcnt.
__device__ inline void async_copy32(unsigned lds_off, const __bf16* gptr) {
    const unsigned long long ga = (unsigned long long)(uintptr_t)gptr;
    asm volatile("global_load_async_to_lds_b128 %0, %1, off\n\t"
                 "global_load_async_to_lds_b128 %0, %1, off offset:16"
                 :
                 : "v"(lds_off), "v"(ga)
                 : "memory");
}

__device__ inline unsigned lds_offset_of(const __bf16* p) {
    return (unsigned)(uintptr_t)(lds_bf16*)p;   // addrspacecast -> raw LDS offset
}

#define WMMA_BF16(ah, bb, c) \
    __builtin_amdgcn_wmma_f32_16x16x32_bf16(false, (ah), false, (bb), (short)0, (c), false, false)

// 24 WMMAs for one K-slice; B fragments pipelined one tile ahead using two
// independent register webs (no rotation copies -> allocator can't collapse
// them into one pair, so ds waits become partial instead of dscnt==0).
__device__ inline void compute_slice(v8f (&acc)[8], const __bf16* sbh, const __bf16* sbl,
                                     const v16bf& a_hi, const v16bf& a_lo) {
    v16bf bh0 = *reinterpret_cast<const v16bf*>(sbh);
    v16bf bl0 = *reinterpret_cast<const v16bf*>(sbl);
#pragma unroll
    for (int nt = 0; nt < 8; nt += 2) {
        const v16bf bh1 = *reinterpret_cast<const v16bf*>(sbh + (nt + 1) * 512);
        const v16bf bl1 = *reinterpret_cast<const v16bf*>(sbl + (nt + 1) * 512);

        acc[nt] = WMMA_BF16(a_hi, bh0, acc[nt]);
        acc[nt] = WMMA_BF16(a_lo, bh0, acc[nt]);
        acc[nt] = WMMA_BF16(a_hi, bl0, acc[nt]);

        if (nt + 2 < 8) {   // reload web 0 for tile nt+2 (after its last use)
            bh0 = *reinterpret_cast<const v16bf*>(sbh + (nt + 2) * 512);
            bl0 = *reinterpret_cast<const v16bf*>(sbl + (nt + 2) * 512);
        }

        acc[nt + 1] = WMMA_BF16(a_hi, bh1, acc[nt + 1]);
        acc[nt + 1] = WMMA_BF16(a_lo, bh1, acc[nt + 1]);
        acc[nt + 1] = WMMA_BF16(a_hi, bl1, acc[nt + 1]);
    }
}

// ---------------------------------------------------------------------------
// Kernel 0: one-time split of W into bf16 hi/lo copies (row-major [OUT_F][IN_F]),
// stored in workspace (2 MB total; L2-resident thereafter).
// ---------------------------------------------------------------------------
__global__ __launch_bounds__(256) void w_split(const float* __restrict__ w,
                                               __bf16* __restrict__ w_hi,
                                               __bf16* __restrict__ w_lo) {
    const int i = blockIdx.x * 256 + threadIdx.x;
    const float  v = w[i];
    const __bf16 h = (__bf16)v;
    w_hi[i] = h;
    w_lo[i] = (__bf16)(v - (float)h);
}

// ---------------------------------------------------------------------------
// Kernel 1: drive = (1 - decay) * (x @ W^T + bias) written directly into the
// states region of d_out at [b, t+1, o].
// Block = 256 threads = 8 waves; block tile 128(M) x 128(N); wave tile 16x128.
// W K-slices staged into double-buffered LDS via GLOBAL_LOAD_ASYNC_TO_LDS_B128,
// overlapped with the current slice's 24 WMMAs; last K-step peeled so the
// steady-state loop has no predication.
// ---------------------------------------------------------------------------
__global__ __launch_bounds__(256) void li_gemm_drive(
    const float*  __restrict__ x,      // [MROWS, IN_F] fp32
    const __bf16* __restrict__ w_hi,   // [OUT_F, IN_F] bf16 (hi)
    const __bf16* __restrict__ w_lo,   // [OUT_F, IN_F] bf16 (lo residual)
    const float*  __restrict__ bias,   // [OUT_F]
    const float*  __restrict__ decay,  // [OUT_F]
    float* __restrict__ states)        // [BATCH, TSTEPS+1, OUT_F]
{
    // [buf][hi/lo][128 rows x 32 k] bf16, row-major, 64 B per row
    __shared__ __align__(32) __bf16 sW[2][2][128 * 32];

    const int lane = threadIdx.x & 31;
    const int wave = threadIdx.x >> 5;      // 0..7
    const int half = lane >> 4;             // 0 or 1
    const int lr   = lane & 15;

    const int m0 = blockIdx.x * 128 + wave * 16;   // wave's row tile
    const int n0 = blockIdx.y * 128;               // block's col range

    const float* __restrict__ ax = x + (size_t)(m0 + lr) * IN_F;

    // staging role: each thread async-copies 32 B of hi and 32 B of lo per slice
    const int srow  = threadIdx.x >> 1;     // 0..127 (row within N slice)
    const int shalf = threadIdx.x & 1;      // which 16-element K half
    const __bf16* __restrict__ gh = w_hi + (size_t)(n0 + srow) * IN_F + shalf * 16;
    const __bf16* __restrict__ gl = w_lo + (size_t)(n0 + srow) * IN_F + shalf * 16;
    const int soff = srow * 32 + shalf * 16;           // element offset in slice
    const unsigned lds_h0 = lds_offset_of(&sW[0][0][soff]);
    const unsigned lds_l0 = lds_offset_of(&sW[0][1][soff]);
    const unsigned lds_h1 = lds_offset_of(&sW[1][0][soff]);
    const unsigned lds_l1 = lds_offset_of(&sW[1][1][soff]);

    const int fbase = lr * 32 + half * 16;             // fragment row/K-half

    v8f acc[8];
#pragma unroll
    for (int i = 0; i < 8; ++i) acc[i] = (v8f){};

    // ---- prologue: slice 0 -> LDS buf 0 (async); A fragment 0 -> regs
    async_copy32(lds_h0, gh);
    async_copy32(lds_l0, gl);
    F16x a_cur;
    {
        const float4* pa = reinterpret_cast<const float4*>(ax + half * 16);
        a_cur.q[0] = pa[0]; a_cur.q[1] = pa[1];   // k + 0..7
        a_cur.q[2] = pa[2]; a_cur.q[3] = pa[3];   // k + 8..15
    }
    asm volatile("s_wait_asynccnt 0x0" ::: "memory");
    __syncthreads();

    // ---- steady state: 31 slices, always prefetching the next one
    int buf = 0;
    for (int kk = 0; kk < IN_F - 32; kk += 32) {
        // issue next slice's async copies + A prefetch (hidden by WMMAs)
        async_copy32(buf ? lds_h0 : lds_h1, gh + kk + 32);
        async_copy32(buf ? lds_l0 : lds_l1, gl + kk + 32);
        F16x a_nxt;
        {
            const float4* pa = reinterpret_cast<const float4*>(ax + kk + 32 + half * 16);
            a_nxt.q[0] = pa[0]; a_nxt.q[1] = pa[1];
            a_nxt.q[2] = pa[2]; a_nxt.q[3] = pa[3];
        }

        // A fragment (16x32 bf16): lanes<16 K {0..15}, lanes>=16 K {16..31}
        v16bf a_hi, a_lo;
        split_bf16(a_cur, a_hi, a_lo);

        compute_slice(acc, &sW[buf][0][fbase], &sW[buf][1][fbase], a_hi, a_lo);

        a_cur = a_nxt;
        buf ^= 1;
        asm volatile("s_wait_asynccnt 0x0" ::: "memory");
        __syncthreads();
    }

    // ---- final slice (no prefetch, no trailing barrier needed before epilogue)
    {
        v16bf a_hi, a_lo;
        split_bf16(a_cur, a_hi, a_lo);
        compute_slice(acc, &sW[buf][0][fbase], &sW[buf][1][fbase], a_hi, a_lo);
    }

    // ---- epilogue: drive = (1 - decay[o]) * (acc + bias[o]) -> states[b,t+1,o]
#pragma unroll
    for (int nt = 0; nt < 8; ++nt) {
        const int   col = n0 + nt * 16 + lr;
        const float sc  = 1.0f - decay[col];
        const float bi  = bias[col];
#pragma unroll
        for (int r = 0; r < 8; ++r) {
            // C/D layout: VGPR r -> M = r (lanes 0-15) or r+8 (lanes 16-31)
            const int    m    = m0 + r + 8 * half;       // global row = b*TSTEPS + t
            const int    b    = m >> 10;                 // TSTEPS = 1024
            const size_t base = (size_t)(m + b + 1) * OUT_F;  // (b*(T+1)+t+1)*OUT_F
            states[base + col] = sc * (acc[nt][r] + bi);
        }
    }
}

// ---------------------------------------------------------------------------
// Kernel 2: in-place leaky-integrator scan over the states region,
// mirroring u_t into outputs. One thread per (b, o); coalesced over o.
// ---------------------------------------------------------------------------
__global__ __launch_bounds__(256) void li_scan(
    const float* __restrict__ decay,
    float* __restrict__ out,          // [BATCH, TSTEPS, OUT_F]
    float* __restrict__ states)       // [BATCH, TSTEPS+1, OUT_F]
{
    const int tid = blockIdx.x * blockDim.x + threadIdx.x;   // 0 .. BATCH*OUT_F-1
    const int b = tid >> 9;                                  // OUT_F = 512
    const int o = tid & (OUT_F - 1);

    const float  dec = decay[o];
    const size_t sb  = (size_t)b * (TSTEPS + 1) * OUT_F + o;
    const size_t ob  = (size_t)b * TSTEPS * OUT_F + o;

    states[sb] = 0.0f;        // u0
    float u = 0.0f;
#pragma unroll 8
    for (int t = 0; t < TSTEPS; ++t) {
        const size_t p = sb + (size_t)(t + 1) * OUT_F;
        const float  d = states[p];     // drive written by kernel 1
        u = fmaf(dec, u, d);
        states[p] = u;                  // states[b, t+1, o] = u_t
        out[ob + (size_t)t * OUT_F] = u;
    }
}

// ---------------------------------------------------------------------------
extern "C" void kernel_launch(void* const* d_in, const int* in_sizes, int n_in,
                              void* d_out, int out_size, void* d_ws, size_t ws_size,
                              hipStream_t stream) {
    (void)in_sizes; (void)n_in; (void)out_size; (void)ws_size;

    const float* x     = (const float*)d_in[0];   // [32,1024,1024]
    const float* w     = (const float*)d_in[1];   // [512,1024]
    const float* bias  = (const float*)d_in[2];   // [512]
    const float* decay = (const float*)d_in[3];   // [512]

    float* out    = (float*)d_out;                            // outputs: B*T*OUT
    float* states = out + (size_t)BATCH * TSTEPS * OUT_F;     // states: B*(T+1)*OUT

    __bf16* w_hi = (__bf16*)d_ws;                 // 1 MB
    __bf16* w_lo = w_hi + WELEMS;                 // 1 MB

    w_split<<<WELEMS / 256, 256, 0, stream>>>(w, w_hi, w_lo);

    dim3 grid(MROWS / 128, OUT_F / 128);          // 256 x 4 blocks, 8 waves each
    li_gemm_drive<<<grid, 256, 0, stream>>>(x, w_hi, w_lo, bias, decay, states);

    li_scan<<<(BATCH * OUT_F) / 256, 256, 0, stream>>>(decay, out, states);
}
